// PeptideTriStreamModel_41515153883230
// MI455X (gfx1250) — compile-verified, matrix operand order checked by reading, not run on hardware
//
#include <hip/hip_runtime.h>
#include <math.h>

// ---------------------------------------------------------------------------
// CDNA5 / gfx1250 implementation of the GVP-GNN reference.
// All dense projections run through v_wmma_f32_16x16x32_f16 (wave32 WMMA,
// fp32 accumulate). Edge pipeline is fully fused: gather -> WMMA -> LN ->
// GeLU -> atomic scatter, with no E x 384 intermediate ever hitting HBM.
// Edge-loop weights (B fragments) are held in registers across tiles.
// ---------------------------------------------------------------------------

typedef _Float16 half_t;
typedef __attribute__((ext_vector_type(16))) _Float16 v16h;
typedef __attribute__((ext_vector_type(8)))  float    v8f;

#define NN   20000
#define EE   640000
#define ESMD 1280
#define HH   128
#define OUTD 256
#define LL   3
#define NGG  64

// K index held by element j of a 16-bit A/B fragment in lane `lane`
// (per CDNA5 ISA 7.12.2: 16-bit A 16x32 layout; B assumed symmetric by column).
__device__ __forceinline__ int frag_k(int j, int lane) {
  return ((j >> 3) << 4) + ((lane >> 4) << 3) + (j & 7);
}

__device__ __forceinline__ v8f wmma_f32_16x16x32(v16h a, v16h b, v8f c) {
  return __builtin_amdgcn_wmma_f32_16x16x32_f16(false, a, false, b, (short)0, c,
                                                false, false);
}

__device__ __forceinline__ float gelu_exact(float x) {
  return 0.5f * x * (1.0f + erff(x * 0.70710678118654752f));
}
__device__ __forceinline__ float sigmoidf_(float x) {
  return 1.0f / (1.0f + expf(-x));
}
__device__ __forceinline__ void norm3(float& x, float& y, float& z) {
  float n2 = x * x + y * y + z * z;
  if (n2 > 0.0f) {
    float s = 1.0f / fmaxf(sqrtf(n2), 1e-12f);
    x *= s; y *= s; z *= s;
  } else {
    x = 0.0f; y = 0.0f; z = 0.0f;
  }
}

// Sum `v` across the 16 lanes of this half-wave (lanes 0-15 or 16-31).
__device__ __forceinline__ float half_wave_sum16(float v) {
  v += __shfl_xor(v, 1, 32);
  v += __shfl_xor(v, 2, 32);
  v += __shfl_xor(v, 4, 32);
  v += __shfl_xor(v, 8, 32);
  return v;
}

// ---------------------------------------------------------------------------
// Repack B = W.T into WMMA fragment order. W is (rows=n, cols=k), row-major,
// row length = rowStride. Output tile (kt,nt) -> 32 lanes x 16 halves.
// grid = ktiles*ntiles blocks of 32 threads.
// ---------------------------------------------------------------------------
__global__ void repack_b_kernel(const float* __restrict__ W, int rowStride,
                                int ntiles, half_t* __restrict__ out) {
  int tile = blockIdx.x;
  int kt = tile / ntiles, nt = tile % ntiles;
  int lane = threadIdx.x;
  int n = nt * 16 + (lane & 15);
  v16h frag;
#pragma unroll
  for (int j = 0; j < 16; ++j) {
    int k = kt * 32 + frag_k(j, lane);
    frag[j] = (half_t)W[(size_t)n * rowStride + k];
  }
  ((v16h*)out)[tile * 32 + lane] = frag;
}

// ---------------------------------------------------------------------------
// node_s = concat(esm, geo[:,6:15]) @ W.T + b   (K = 1280 via WMMA + 9 scalar)
// ---------------------------------------------------------------------------
__global__ void __launch_bounds__(256) node_proj_kernel(
    const float* __restrict__ esm, const float* __restrict__ geo,
    const float* __restrict__ W, const float* __restrict__ bias,
    const half_t* __restrict__ Wp, float* __restrict__ node_s) {
  __shared__ half_t Ah[16 * ESMD];  // 40 KB
  const int tid = threadIdx.x, wave = tid >> 5, lane = tid & 31;
  const int m = lane & 15;
  const int row0 = blockIdx.x * 16;
  for (int idx = tid; idx < 16 * ESMD; idx += 256) {
    int r = idx / ESMD, k = idx - r * ESMD;
    Ah[idx] = (half_t)esm[(size_t)(row0 + r) * ESMD + k];
  }
  __syncthreads();
  v8f acc = {};
  for (int kt = 0; kt < ESMD / 32; ++kt) {
    v16h a;
#pragma unroll
    for (int j = 0; j < 16; ++j) a[j] = Ah[m * ESMD + kt * 32 + frag_k(j, lane)];
    v16h b = ((const v16h*)Wp)[(kt * 8 + wave) * 32 + lane];
    acc = wmma_f32_16x16x32(a, b, acc);
  }
  const int n = wave * 16 + m;
  const float bn = bias[n];
#pragma unroll
  for (int r = 0; r < 8; ++r) {
    int node = row0 + ((lane >> 4) << 3) + r;
    float v = acc[r] + bn;
#pragma unroll
    for (int j = 0; j < 9; ++j)
      v += geo[(size_t)node * 15 + 6 + j] * W[(size_t)n * 1289 + 1280 + j];
    node_s[(size_t)node * HH + n] = v;
  }
}

// ---------------------------------------------------------------------------
// node_v[i] = [ca_cb, d, normal, t]  (4 x 3)
// ---------------------------------------------------------------------------
__global__ void node_vec_kernel(const float* __restrict__ coords,
                                const float* __restrict__ geo,
                                float* __restrict__ node_v) {
  int i = blockIdx.x * blockDim.x + threadIdx.x;
  if (i >= NN) return;
  float cx = coords[i * 3 + 0], cy = coords[i * 3 + 1], cz = coords[i * 3 + 2];
  float ax = geo[(size_t)i * 15 + 3] - cx;
  float ay = geo[(size_t)i * 15 + 4] - cy;
  float az = geo[(size_t)i * 15 + 5] - cz;
  norm3(ax, ay, az);
  float dx = 0, dy = 0, dz = 0;
  if (i < NN - 1) {
    dx = coords[(i + 1) * 3 + 0] - cx;
    dy = coords[(i + 1) * 3 + 1] - cy;
    dz = coords[(i + 1) * 3 + 2] - cz;
  }
  norm3(dx, dy, dz);
  float nx = ay * dz - az * dy, ny = az * dx - ax * dz, nz = ax * dy - ay * dx;
  norm3(nx, ny, nz);
  float tx = 0, ty = 0, tz = 0;
  if (i > 0 && i < NN - 1) {
    tx = coords[(i + 1) * 3 + 0] - coords[(i - 1) * 3 + 0];
    ty = coords[(i + 1) * 3 + 1] - coords[(i - 1) * 3 + 1];
    tz = coords[(i + 1) * 3 + 2] - coords[(i - 1) * 3 + 2];
  }
  norm3(tx, ty, tz);
  float* o = node_v + (size_t)i * 12;
  o[0] = ax; o[1] = ay; o[2]  = az;
  o[3] = dx; o[4] = dy; o[5]  = dz;
  o[6] = nx; o[7] = ny; o[8]  = nz;
  o[9] = tx; o[10] = ty; o[11] = tz;
}

// ---------------------------------------------------------------------------
// Fused edge-message GVP + scatter-add. 16 edges per tile, grid-stride.
// Each wave keeps its 12 Wss B-fragments in registers across all tiles.
// ---------------------------------------------------------------------------
__global__ void __launch_bounds__(256) msg_kernel(
    const float* __restrict__ node_s, const float* __restrict__ node_v,
    const float* __restrict__ eattr, const int* __restrict__ src,
    const int* __restrict__ dst, const half_t* __restrict__ WssP,
    const float* __restrict__ Wvs, const float* __restrict__ Wsv,
    const float* __restrict__ Wvv, const float* __restrict__ lng,
    const float* __restrict__ lnb, const float* __restrict__ eW,
    const float* __restrict__ eb, float* __restrict__ ag_s,
    float* __restrict__ ag_v) {
  __shared__ half_t ms[16 * 384];
  __shared__ float mv[16][9][3];
  __shared__ float vn[16][9];
  __shared__ float redS[16];
  __shared__ float redQ[16];
  const int tid = threadIdx.x, wave = tid >> 5, lane = tid & 31;
  const int m = lane & 15;
  const int n = wave * 16 + m;
  float wv[9];
#pragma unroll
  for (int k = 0; k < 9; ++k) wv[k] = Wvs[n * 9 + k];
  const float g = lng[n], bln = lnb[n];
  // Hoist this wave's 12 B fragments (K=384) into registers.
  v16h bfr[12];
#pragma unroll
  for (int kt = 0; kt < 12; ++kt)
    bfr[kt] = ((const v16h*)WssP)[(kt * 8 + wave) * 32 + lane];

  for (int tile = blockIdx.x; tile < EE / 16; tile += gridDim.x) {
    const int e0 = tile * 16;
    // Prefetch next grid-stride tile's edge stream (global_prefetch_b8).
    {
      int ne0 = e0 + 16 * (int)gridDim.x;
      if (ne0 < EE) {
        __builtin_prefetch(&src[ne0], 0, 0);
        __builtin_prefetch(&dst[ne0], 0, 0);
        __builtin_prefetch(&eattr[(size_t)ne0 * 16], 0, 0);
      }
    }
    // Stage scalar message input [s_src | s_dst | edge_s(on the fly)] as f16.
    for (int idx = tid; idx < 16 * 384; idx += 256) {
      int e = idx / 384, k = idx - e * 384;
      int ge = e0 + e;
      float v;
      if (k < 128) {
        v = node_s[(size_t)src[ge] * HH + k];
      } else if (k < 256) {
        v = node_s[(size_t)dst[ge] * HH + (k - 128)];
      } else {
        int c = k - 256;
        const float* a = eattr + (size_t)ge * 16;
        const float* w = eW + c * 13;
        float s = eb[c] + a[0] * w[0] + a[1] * w[1];
#pragma unroll
        for (int j = 0; j < 11; ++j) s += a[5 + j] * w[2 + j];
        v = s;
      }
      ms[idx] = (half_t)v;
    }
    // Stage vector channels + norms.
    for (int idx = tid; idx < 16 * 9; idx += 256) {
      int e = idx / 9, i = idx - e * 9;
      int ge = e0 + e;
      float x, y, z;
      if (i < 4) {
        const float* pv = node_v + (size_t)src[ge] * 12 + i * 3;
        x = pv[0]; y = pv[1]; z = pv[2];
      } else if (i < 8) {
        const float* pv = node_v + (size_t)dst[ge] * 12 + (i - 4) * 3;
        x = pv[0]; y = pv[1]; z = pv[2];
      } else {
        const float* a = eattr + (size_t)ge * 16;
        x = a[2]; y = a[3]; z = a[4];
      }
      mv[e][i][0] = x; mv[e][i][1] = y; mv[e][i][2] = z;
      float n2 = x * x + y * y + z * z;
      vn[e][i] = (n2 > 0.0f) ? sqrtf(n2) : 0.0f;
    }
    if (tid < 16) { redS[tid] = 0.0f; redQ[tid] = 0.0f; }
    __syncthreads();

    // 16x128 = 8 waves x (16x16 tile), K = 384 in 12 WMMA steps.
    v8f acc = {};
#pragma unroll
    for (int kt = 0; kt < 12; ++kt) {
      v16h a;
#pragma unroll
      for (int j = 0; j < 16; ++j)
        a[j] = ms[m * 384 + kt * 32 + frag_k(j, lane)];
      acc = wmma_f32_16x16x32(a, bfr[kt], acc);
    }
    // Add vn @ Wvs.T, then LN statistics: butterfly-reduce over the 16 lanes
    // of each half-wave (those lanes hold the 16 columns of the same rows),
    // leaving 2 LDS atomics per row per wave.
#pragma unroll
    for (int r = 0; r < 8; ++r) {
      int e = ((lane >> 4) << 3) + r;
      float s = 0.0f;
#pragma unroll
      for (int k = 0; k < 9; ++k) s += vn[e][k] * wv[k];
      acc[r] += s;
      float ps = half_wave_sum16(acc[r]);
      float pq = half_wave_sum16(acc[r] * acc[r]);
      if (m == 0) {
        atomicAdd(&redS[e], ps);
        atomicAdd(&redQ[e], pq);
      }
    }
    __syncthreads();
#pragma unroll
    for (int r = 0; r < 8; ++r) {
      int e = ((lane >> 4) << 3) + r;
      float mean = redS[e] * (1.0f / 128.0f);
      float var = redQ[e] * (1.0f / 128.0f) - mean * mean;
      float x = (acc[r] - mean) * rsqrtf(var + 1e-5f) * g + bln;
      atomicAdd(&ag_s[(size_t)dst[e0 + e] * HH + n], gelu_exact(x));
    }
    // Gated vector path (4 outputs / edge).
    if (tid < 64) {
      int e = tid >> 2, o = tid & 3;
      float dot = 0.0f;
      for (int k = 0; k < 384; ++k)
        dot += (float)ms[e * 384 + k] * Wsv[o * 384 + k];
      float gate = sigmoidf_(dot);
      int de = dst[e0 + e];
#pragma unroll
      for (int c = 0; c < 3; ++c) {
        float vo = 0.0f;
#pragma unroll
        for (int i = 0; i < 9; ++i) vo += mv[e][i][c] * Wvv[o * 9 + i];
        atomicAdd(&ag_v[(size_t)de * 12 + o * 3 + c], gate * vo);
      }
    }
    __syncthreads();
  }
}

// ---------------------------------------------------------------------------
// Node-update GVP: node_s += ns, node_v += nv. 16 nodes per block.
// ---------------------------------------------------------------------------
__global__ void __launch_bounds__(256) upd_kernel(
    float* __restrict__ node_s, float* __restrict__ node_v,
    const float* __restrict__ ag_s, const float* __restrict__ ag_v,
    const half_t* __restrict__ WssP, const float* __restrict__ Wvs,
    const float* __restrict__ Wsv, const float* __restrict__ Wvv,
    const float* __restrict__ lng, const float* __restrict__ lnb) {
  __shared__ half_t us[16 * 256];
  __shared__ float uv[16][8][3];
  __shared__ float vn[16][8];
  __shared__ float redS[16];
  __shared__ float redQ[16];
  const int tid = threadIdx.x, wave = tid >> 5, lane = tid & 31;
  const int m = lane & 15;
  const int n = wave * 16 + m;
  const int r0 = blockIdx.x * 16;
  for (int idx = tid; idx < 16 * 256; idx += 256) {
    int e = idx >> 8, k = idx & 255;
    int node = r0 + e;
    float v = (k < 128) ? node_s[(size_t)node * HH + k]
                        : ag_s[(size_t)node * HH + (k - 128)];
    us[idx] = (half_t)v;
  }
  for (int idx = tid; idx < 16 * 8; idx += 256) {
    int e = idx >> 3, i = idx & 7;
    int node = r0 + e;
    const float* pv = (i < 4) ? (node_v + (size_t)node * 12 + i * 3)
                              : (ag_v + (size_t)node * 12 + (i - 4) * 3);
    float x = pv[0], y = pv[1], z = pv[2];
    uv[e][i][0] = x; uv[e][i][1] = y; uv[e][i][2] = z;
    float n2 = x * x + y * y + z * z;
    vn[e][i] = (n2 > 0.0f) ? sqrtf(n2) : 0.0f;
  }
  if (tid < 16) { redS[tid] = 0.0f; redQ[tid] = 0.0f; }
  __syncthreads();
  v8f acc = {};
#pragma unroll
  for (int kt = 0; kt < 8; ++kt) {
    v16h a;
#pragma unroll
    for (int j = 0; j < 16; ++j) a[j] = us[m * 256 + kt * 32 + frag_k(j, lane)];
    v16h b = ((const v16h*)WssP)[(kt * 8 + wave) * 32 + lane];
    acc = wmma_f32_16x16x32(a, b, acc);
  }
  float wv[8];
#pragma unroll
  for (int k = 0; k < 8; ++k) wv[k] = Wvs[n * 8 + k];
#pragma unroll
  for (int r = 0; r < 8; ++r) {
    int e = ((lane >> 4) << 3) + r;
    float s = 0.0f;
#pragma unroll
    for (int k = 0; k < 8; ++k) s += vn[e][k] * wv[k];
    acc[r] += s;
    float ps = half_wave_sum16(acc[r]);
    float pq = half_wave_sum16(acc[r] * acc[r]);
    if (m == 0) {
      atomicAdd(&redS[e], ps);
      atomicAdd(&redQ[e], pq);
    }
  }
  __syncthreads();
  const float g = lng[n], bln = lnb[n];
#pragma unroll
  for (int r = 0; r < 8; ++r) {
    int e = ((lane >> 4) << 3) + r;
    float mean = redS[e] * (1.0f / 128.0f);
    float var = redQ[e] * (1.0f / 128.0f) - mean * mean;
    float x = (acc[r] - mean) * rsqrtf(var + 1e-5f) * g + bln;
    node_s[(size_t)(r0 + e) * HH + n] += gelu_exact(x);
  }
  if (tid < 64) {
    int e = tid >> 2, o = tid & 3;
    float dot = 0.0f;
    for (int k = 0; k < 256; ++k)
      dot += (float)us[e * 256 + k] * Wsv[o * 256 + k];
    float gate = sigmoidf_(dot);
#pragma unroll
    for (int c = 0; c < 3; ++c) {
      float vo = 0.0f;
#pragma unroll
      for (int i = 0; i < 8; ++i) vo += uv[e][i][c] * Wvv[o * 8 + i];
      node_v[(size_t)(r0 + e) * 12 + o * 3 + c] += gate * vo;
    }
  }
}

// ---------------------------------------------------------------------------
// updated_nodes = node_s @ node_out_W.T + b   (K=128, 80 column tiles)
// ---------------------------------------------------------------------------
__global__ void __launch_bounds__(256) node_out_kernel(
    const float* __restrict__ node_s, const half_t* __restrict__ Wp,
    const float* __restrict__ bias, float* __restrict__ out) {
  __shared__ half_t Ah[16 * 128];
  const int tid = threadIdx.x, wave = tid >> 5, lane = tid & 31;
  const int m = lane & 15;
  const int r0 = blockIdx.x * 16;
  for (int idx = tid; idx < 2048; idx += 256)
    Ah[idx] = (half_t)node_s[(size_t)(r0 + (idx >> 7)) * HH + (idx & 127)];
  __syncthreads();
  for (int ct = wave; ct < 80; ct += 8) {
    v8f acc = {};
#pragma unroll
    for (int kt = 0; kt < 4; ++kt) {
      v16h a;
#pragma unroll
      for (int j = 0; j < 16; ++j)
        a[j] = Ah[m * 128 + kt * 32 + frag_k(j, lane)];
      v16h b = ((const v16h*)Wp)[(kt * 80 + ct) * 32 + lane];
      acc = wmma_f32_16x16x32(a, b, acc);
    }
    int n = ct * 16 + m;
    float bn = bias[n];
#pragma unroll
    for (int r = 0; r < 8; ++r) {
      int node = r0 + ((lane >> 4) << 3) + r;
      out[(size_t)node * ESMD + n] = acc[r] + bn;
    }
  }
}

// ---------------------------------------------------------------------------
// graph feature projection + segment-sum via atomics into NG x 256 accum.
// ---------------------------------------------------------------------------
__global__ void __launch_bounds__(256) graph_out_kernel(
    const float* __restrict__ node_s, const half_t* __restrict__ Wp,
    const float* __restrict__ bias, const int* __restrict__ batch,
    float* __restrict__ accum) {
  __shared__ half_t Ah[16 * 128];
  const int tid = threadIdx.x, wave = tid >> 5, lane = tid & 31;
  const int m = lane & 15;
  const int r0 = blockIdx.x * 16;
  for (int idx = tid; idx < 2048; idx += 256)
    Ah[idx] = (half_t)node_s[(size_t)(r0 + (idx >> 7)) * HH + (idx & 127)];
  __syncthreads();
  for (int ct = wave; ct < 16; ct += 8) {
    v8f acc = {};
#pragma unroll
    for (int kt = 0; kt < 4; ++kt) {
      v16h a;
#pragma unroll
      for (int j = 0; j < 16; ++j)
        a[j] = Ah[m * 128 + kt * 32 + frag_k(j, lane)];
      v16h b = ((const v16h*)Wp)[(kt * 16 + ct) * 32 + lane];
      acc = wmma_f32_16x16x32(a, b, acc);
    }
    int n = ct * 16 + m;
    float bn = bias[n];
#pragma unroll
    for (int r = 0; r < 8; ++r) {
      int node = r0 + ((lane >> 4) << 3) + r;
      atomicAdd(&accum[(size_t)batch[node] * OUTD + n], acc[r] + bn);
    }
  }
}

__global__ void count_kernel(const int* __restrict__ batch,
                             float* __restrict__ cnt) {
  int i = blockIdx.x * blockDim.x + threadIdx.x;
  if (i < NN) atomicAdd(&cnt[batch[i]], 1.0f);
}

__global__ void finalize_kernel(const float* __restrict__ accum,
                                const float* __restrict__ cnt,
                                float* __restrict__ out) {
  int g = blockIdx.x, c = threadIdx.x;
  out[g * OUTD + c] = accum[g * OUTD + c] / fmaxf(cnt[g], 1.0f);
}

// ---------------------------------------------------------------------------
extern "C" void kernel_launch(void* const* d_in, const int* in_sizes, int n_in,
                              void* d_out, int out_size, void* d_ws,
                              size_t ws_size, hipStream_t stream) {
  (void)in_sizes; (void)n_in; (void)out_size; (void)ws_size;
  const float* esm    = (const float*)d_in[0];
  const float* geo    = (const float*)d_in[1];
  const float* coords = (const float*)d_in[2];
  const int*   eidx   = (const int*)d_in[3];
  const float* eattr  = (const float*)d_in[4];
  const int*   batch  = (const int*)d_in[5];
  const float* npW = (const float*)d_in[6];
  const float* npB = (const float*)d_in[7];
  const float* epW = (const float*)d_in[8];
  const float* epB = (const float*)d_in[9];
  const float* mWss = (const float*)d_in[10];
  const float* mWvs = (const float*)d_in[11];
  const float* mWsv = (const float*)d_in[12];
  const float* mWvv = (const float*)d_in[13];
  const float* mLg  = (const float*)d_in[14];
  const float* mLb  = (const float*)d_in[15];
  const float* uWss = (const float*)d_in[16];
  const float* uWvs = (const float*)d_in[17];
  const float* uWsv = (const float*)d_in[18];
  const float* uWvv = (const float*)d_in[19];
  const float* uLg  = (const float*)d_in[20];
  const float* uLb  = (const float*)d_in[21];
  const float* noW  = (const float*)d_in[22];
  const float* noB  = (const float*)d_in[23];
  const float* goW  = (const float*)d_in[24];
  const float* goB  = (const float*)d_in[25];

  float* out_graph = (float*)d_out;                       // 64 x 256
  float* out_nodes = (float*)d_out + (size_t)NGG * OUTD;  // N x 1280

  char* p = (char*)d_ws;
  auto carve = [&](size_t bytes) -> char* {
    char* r = p;
    p += (bytes + 255) & ~(size_t)255;
    return r;
  };
  float* node_s = (float*)carve(sizeof(float) * NN * HH);
  float* node_v = (float*)carve(sizeof(float) * NN * 12);
  float* ag_s   = (float*)carve(sizeof(float) * NN * HH);
  float* ag_v   = (float*)carve(sizeof(float) * NN * 12);
  float* accum  = (float*)carve(sizeof(float) * NGG * OUTD);
  float* cnt    = (float*)carve(sizeof(float) * NGG);
  half_t* pNP = (half_t*)carve(sizeof(half_t) * 40 * 8 * 512);
  half_t* pMsg[LL];
  for (int l = 0; l < LL; ++l) pMsg[l] = (half_t*)carve(sizeof(half_t) * 12 * 8 * 512);
  half_t* pUpd[LL];
  for (int l = 0; l < LL; ++l) pUpd[l] = (half_t*)carve(sizeof(half_t) * 8 * 8 * 512);
  half_t* pNO = (half_t*)carve(sizeof(half_t) * 4 * 80 * 512);
  half_t* pGO = (half_t*)carve(sizeof(half_t) * 4 * 16 * 512);

  // Repack all GEMM weights into WMMA B-fragment order (f16).
  repack_b_kernel<<<40 * 8, 32, 0, stream>>>(npW, 1289, 8, pNP);
  for (int l = 0; l < LL; ++l)
    repack_b_kernel<<<12 * 8, 32, 0, stream>>>(mWss + (size_t)l * HH * 384, 384, 8, pMsg[l]);
  for (int l = 0; l < LL; ++l)
    repack_b_kernel<<<8 * 8, 32, 0, stream>>>(uWss + (size_t)l * HH * 256, 256, 8, pUpd[l]);
  repack_b_kernel<<<4 * 80, 32, 0, stream>>>(noW, 128, 80, pNO);
  repack_b_kernel<<<4 * 16, 32, 0, stream>>>(goW, 128, 16, pGO);

  node_proj_kernel<<<NN / 16, 256, 0, stream>>>(esm, geo, npW, npB, pNP, node_s);
  node_vec_kernel<<<(NN + 255) / 256, 256, 0, stream>>>(coords, geo, node_v);

  const int* srcp = eidx;
  const int* dstp = eidx + EE;
  for (int l = 0; l < LL; ++l) {
    hipMemsetAsync(ag_s, 0, sizeof(float) * NN * HH, stream);
    hipMemsetAsync(ag_v, 0, sizeof(float) * NN * 12, stream);
    msg_kernel<<<4096, 256, 0, stream>>>(
        node_s, node_v, eattr, srcp, dstp, pMsg[l],
        mWvs + (size_t)l * HH * 9, mWsv + (size_t)l * 4 * 384,
        mWvv + (size_t)l * 4 * 9, mLg + l * HH, mLb + l * HH, epW, epB,
        ag_s, ag_v);
    upd_kernel<<<NN / 16, 256, 0, stream>>>(
        node_s, node_v, ag_s, ag_v, pUpd[l],
        uWvs + (size_t)l * HH * 8, uWsv + (size_t)l * 4 * 256,
        uWvv + (size_t)l * 4 * 8, uLg + l * HH, uLb + l * HH);
  }

  node_out_kernel<<<NN / 16, 256, 0, stream>>>(node_s, pNO, noB, out_nodes);
  hipMemsetAsync(accum, 0, sizeof(float) * NGG * OUTD, stream);
  hipMemsetAsync(cnt, 0, sizeof(float) * NGG, stream);
  graph_out_kernel<<<NN / 16, 256, 0, stream>>>(node_s, pGO, goB, batch, accum);
  count_kernel<<<(NN + 255) / 256, 256, 0, stream>>>(batch, cnt);
  finalize_kernel<<<NGG, OUTD, 0, stream>>>(accum, cnt, out_graph);
}
